// AttentionBlock_7791070674947
// MI455X (gfx1250) — matrix operand order repaired and checked by reading, not compiled, not run on hardware
//
#include <hip/hip_runtime.h>
#include <hip/hip_bf16.h>

typedef __attribute__((ext_vector_type(16))) __bf16 v16bf;
typedef __attribute__((ext_vector_type(8)))  float  v8f;
typedef int v4i_vec __attribute__((vector_size(16)));

union Frag16 {
    v16bf  v;
    uint4  q[2];
    __bf16 e[16];
};

__device__ __forceinline__ v8f wmma_bf16(const v16bf& a, const v16bf& b, const v8f& c) {
    return __builtin_amdgcn_wmma_f32_16x16x32_bf16(false, a, false, b, (short)0, c, false, false);
}

// ---------------------------------------------------------------------------
// Async global->LDS copy path (GLOBAL_LOAD_ASYNC_TO_LDS_B128, ASYNCcnt).
// Falls back to sync load+ds_store if the builtins are not available.
// ---------------------------------------------------------------------------
#if defined(__has_builtin)
#  if __has_builtin(__builtin_amdgcn_global_load_async_to_lds_b128) && \
      __has_builtin(__builtin_amdgcn_s_wait_asynccnt)
#    define USE_ASYNC_LDS 1
#  endif
#endif
#ifndef USE_ASYNC_LDS
#  define USE_ASYNC_LDS 0
#endif

__device__ __forceinline__ void cp16(void* lds, const void* g) {
#if USE_ASYNC_LDS
    __builtin_amdgcn_global_load_async_to_lds_b128(
        (v4i_vec*)g, (v4i_vec*)lds, 0, 0);
#else
    *(uint4*)lds = *(const uint4*)g;
#endif
}

template <int N>
__device__ __forceinline__ void cp_wait() {
#if USE_ASYNC_LDS
    __builtin_amdgcn_s_wait_asynccnt(N);
#endif
}

// ---------------------------------------------------------------------------
// Prep kernels: fp32 -> bf16 convert, and convert+transpose for weights.
// ---------------------------------------------------------------------------
__global__ void k_cvt_bf16(const float* __restrict__ in, __bf16* __restrict__ out, int n) {
    int i = blockIdx.x * blockDim.x + threadIdx.x;
    if (i < n) out[i] = (__bf16)in[i];
}

// in: [R][Cc] f32 row-major -> out: [Cc][R] bf16 row-major
__global__ void k_transpose_bf16(const float* __restrict__ in, __bf16* __restrict__ out,
                                 int R, int Cc) {
    int i = blockIdx.x * blockDim.x + threadIdx.x;
    if (i >= R * Cc) return;
    int r = i / Cc, c = i - r * Cc;
    out[(size_t)c * R + r] = (__bf16)in[i];
}

// ---------------------------------------------------------------------------
// GEMM 1: X[8192,256] x Wqkv -> scatter into Q,K [b,h,n,256] and Vt [b,h,256,n]
// Bt is Wqkv transposed: [3072][256] bf16.
// Workgroup tile 128x128, 8 waves of 64x32, K-step 32, double-buffered LDS.
// ---------------------------------------------------------------------------
#define GP 40   // LDS row pitch in bf16 (80B = 20 banks -> conflict-free 16B frags)

__global__ __launch_bounds__(256) void k_gemm_qkv(
        const __bf16* __restrict__ A, const __bf16* __restrict__ Bt,
        __bf16* __restrict__ Qp, __bf16* __restrict__ Kp, __bf16* __restrict__ Vtp) {
    constexpr int BM = 128, BN = 128, KD = 256, STEPS = KD / 32;
    __shared__ __bf16 As[2][BM][GP];
    __shared__ __bf16 Bs[2][BN][GP];

    const int tid  = threadIdx.x;
    const int lane = tid & 31, wave = tid >> 5;
    const int lh = lane & 15, hsel = lane >> 4;
    const int m0 = blockIdx.x * BM, n0 = blockIdx.y * BN;
    const int wm = wave >> 2, wn = wave & 3;     // 2x4 waves, wave tile 64x32

    auto issue_tiles = [&](int k0, int buf) {
#pragma unroll
        for (int i = 0; i < (BM * 4) / 256; ++i) {       // A tile: BM x 32
            int q = tid + i * 256;
            int row = q >> 2, cc = q & 3;
            cp16(&As[buf][row][cc * 8], A + (size_t)(m0 + row) * KD + k0 + cc * 8);
        }
#pragma unroll
        for (int i = 0; i < (BN * 4) / 256; ++i) {       // B tile: BN x 32 (n-major)
            int q = tid + i * 256;
            int row = q >> 2, cc = q & 3;
            cp16(&Bs[buf][row][cc * 8], Bt + (size_t)(n0 + row) * KD + k0 + cc * 8);
        }
    };

    const v8f vz = {0.f,0.f,0.f,0.f,0.f,0.f,0.f,0.f};
    v8f acc[4][2];
#pragma unroll
    for (int i = 0; i < 4; ++i)
#pragma unroll
        for (int j = 0; j < 2; ++j) acc[i][j] = vz;

    issue_tiles(0, 0);
    for (int s = 0; s < STEPS; ++s) {
        const int cur = s & 1;
        if (s + 1 < STEPS) {
            issue_tiles((s + 1) * 32, (s + 1) & 1);
            cp_wait<4>();               // drain older group, keep new 4 in flight
        } else {
            cp_wait<0>();
        }
        __syncthreads();

        Frag16 af[4];
#pragma unroll
        for (int mt = 0; mt < 4; ++mt) {
            const __bf16* p = &As[cur][wm * 64 + mt * 16 + lh][hsel * 8];
            af[mt].q[0] = *(const uint4*)p;
            af[mt].q[1] = *(const uint4*)(p + 16);
        }
#pragma unroll
        for (int nt = 0; nt < 2; ++nt) {
            Frag16 bfr;
            const __bf16* p = &Bs[cur][wn * 32 + nt * 16 + lh][hsel * 8];
            bfr.q[0] = *(const uint4*)p;
            bfr.q[1] = *(const uint4*)(p + 16);
#pragma unroll
            for (int mt = 0; mt < 4; ++mt)
                acc[mt][nt] = wmma_bf16(af[mt].v, bfr.v, acc[mt][nt]);
        }
        __syncthreads();
    }

    // Epilogue: scatter to Q / K / V^T in bf16
#pragma unroll
    for (int mt = 0; mt < 4; ++mt)
#pragma unroll
        for (int nt = 0; nt < 2; ++nt)
#pragma unroll
            for (int r = 0; r < 8; ++r) {
                int m = m0 + wm * 64 + mt * 16 + r + hsel * 8;
                int d = n0 + wn * 32 + nt * 16 + lh;
                int head = d / 768, w = d - head * 768;
                int b = m >> 10, ns = m & 1023;
                int bh = b * 4 + head;
                __bf16 bv = (__bf16)acc[mt][nt][r];
                if (w < 256)
                    Qp[(size_t)(bh * 1024 + ns) * 256 + w] = bv;
                else if (w < 512)
                    Kp[(size_t)(bh * 1024 + ns) * 256 + (w - 256)] = bv;
                else
                    Vtp[(size_t)(bh * 256 + (w - 512)) * 1024 + ns] = bv;
            }
}

// ---------------------------------------------------------------------------
// Flash-style attention: each wave owns 16 query rows, WG = 128 rows for one
// (batch, head). K/V streamed in 32-key blocks through double-buffered LDS.
// ---------------------------------------------------------------------------
__global__ __launch_bounds__(256) void k_attention(
        const __bf16* __restrict__ Qp, const __bf16* __restrict__ Kp,
        const __bf16* __restrict__ Vtp, __bf16* __restrict__ Rp) {
    __shared__ __bf16 Kt[2][32][264];   // [key][d]  pitch 528B (bank stride 4)
    __shared__ __bf16 Vs[2][256][40];   // [d][key]  pitch 80B  (bank stride 20)
    __shared__ float  sb[8][16][33];    // per-wave score block, pitch 33
    __shared__ float  rowscale[8][16];
    __shared__ float  rowsum[8][16];

    const int tid  = threadIdx.x;
    const int lane = tid & 31, wave = tid >> 5;
    const int lh = lane & 15, hsel = lane >> 4;
    const int head = blockIdx.y, b = blockIdx.z;
    const int bh = b * 4 + head;
    const int q0 = blockIdx.x * 128 + wave * 16;

    auto issue_tiles = [&](int jb, int buf) {
#pragma unroll
        for (int i = 0; i < 4; ++i) {    // K tile: 32 keys x 256d
            int q = tid + i * 256;
            int row = q >> 5, cc = q & 31;
            cp16(&Kt[buf][row][cc * 8],
                 Kp + (size_t)(bh * 1024 + jb * 32 + row) * 256 + cc * 8);
        }
        const __bf16* vsrc = Vtp + (size_t)(bh * 256 + tid) * 1024 + jb * 32;
#pragma unroll
        for (int i = 0; i < 4; ++i)      // V tile (transposed): 256d x 32 keys
            cp16(&Vs[buf][tid][i * 8], vsrc + i * 8);
    };

    // Q rows held as 8 A-fragments (d = 256 = 8 * K32)
    Frag16 qf[8];
    const __bf16* qbase = Qp + (size_t)(bh * 1024 + q0 + lh) * 256;
#pragma unroll
    for (int kf = 0; kf < 8; ++kf) {
        qf[kf].q[0] = *(const uint4*)(qbase + kf * 32 + hsel * 8);
        qf[kf].q[1] = *(const uint4*)(qbase + kf * 32 + hsel * 8 + 16);
    }

    const v8f vz = {0.f,0.f,0.f,0.f,0.f,0.f,0.f,0.f};
    v8f o[16];
#pragma unroll
    for (int ct = 0; ct < 16; ++ct) o[ct] = vz;
    float m_run = -1e30f, l_run = 0.f;

    issue_tiles(0, 0);
    for (int jb = 0; jb < 32; ++jb) {
        const int cur = jb & 1;
        if (jb + 1 < 32) {
            issue_tiles(jb + 1, (jb + 1) & 1);
            cp_wait<8>();               // drain older K/V group, keep new 8 in flight
        } else {
            cp_wait<0>();
        }
        __syncthreads();

        // S = Q * K^T  (16 x 32 per wave)
        v8f s0 = vz, s1 = vz;
#pragma unroll
        for (int kk = 0; kk < 8; ++kk) {
            Frag16 b0, b1;
            const __bf16* p0 = &Kt[cur][lh][kk * 32 + hsel * 8];
            b0.q[0] = *(const uint4*)p0;  b0.q[1] = *(const uint4*)(p0 + 16);
            const __bf16* p1 = &Kt[cur][16 + lh][kk * 32 + hsel * 8];
            b1.q[0] = *(const uint4*)p1;  b1.q[1] = *(const uint4*)(p1 + 16);
            s0 = wmma_bf16(qf[kk].v, b0.v, s0);
            s1 = wmma_bf16(qf[kk].v, b1.v, s1);
        }

        // stage scaled scores to per-wave LDS block
#pragma unroll
        for (int r = 0; r < 8; ++r) {
            sb[wave][r + hsel * 8][lh]      = s0[r] * 0.0625f;
            sb[wave][r + hsel * 8][16 + lh] = s1[r] * 0.0625f;
        }
        __syncthreads();

        // online softmax: lanes 0..15 each own one query row
        if (lane < 16) {
            float mx = m_run;
#pragma unroll
            for (int j = 0; j < 32; ++j) mx = fmaxf(mx, sb[wave][lane][j]);
            float corr = __expf(m_run - mx);
            float s = l_run * corr;
#pragma unroll
            for (int j = 0; j < 32; ++j) {
                float p = __expf(sb[wave][lane][j] - mx);
                sb[wave][lane][j] = p;
                s += p;
            }
            m_run = mx; l_run = s;
            rowscale[wave][lane] = corr;
        }
        __syncthreads();

        // rescale running output accumulators
        float f[8];
#pragma unroll
        for (int r = 0; r < 8; ++r) f[r] = rowscale[wave][r + hsel * 8];
#pragma unroll
        for (int ct = 0; ct < 16; ++ct)
#pragma unroll
            for (int r = 0; r < 8; ++r) o[ct][r] *= f[r];

        // P fragment (A-matrix layout) from probabilities
        Frag16 pf;
#pragma unroll
        for (int e = 0; e < 16; ++e) {
            int j = (e & 7) + hsel * 8 + (e >> 3) * 16;
            pf.e[e] = (__bf16)sb[wave][lh][j];
        }

        // O += P * V
#pragma unroll
        for (int ct = 0; ct < 16; ++ct) {
            Frag16 vf;
            const __bf16* p = &Vs[cur][ct * 16 + lh][hsel * 8];
            vf.q[0] = *(const uint4*)p;  vf.q[1] = *(const uint4*)(p + 16);
            o[ct] = wmma_bf16(pf.v, vf.v, o[ct]);
        }
        __syncthreads();
    }

    // normalize and write res in [b, n, head*256 + c] bf16 layout
    if (lane < 16) rowsum[wave][lane] = l_run;
    __syncthreads();
    float inv[8];
#pragma unroll
    for (int r = 0; r < 8; ++r) inv[r] = 1.0f / rowsum[wave][r + hsel * 8];
#pragma unroll
    for (int ct = 0; ct < 16; ++ct)
#pragma unroll
        for (int r = 0; r < 8; ++r) {
            int ns = q0 + r + hsel * 8;
            int c  = ct * 16 + lh;
            Rp[(size_t)(b * 1024 + ns) * 1024 + head * 256 + c] =
                (__bf16)(o[ct][r] * inv[r]);
        }
}

// ---------------------------------------------------------------------------
// GEMM 3: R[8192,1024] x Wout^T[256,1024] + b_out + x  -> out f32 [8192,256]
// Workgroup tile 128x64, 8 waves of 32x32, double-buffered LDS.
// ---------------------------------------------------------------------------
__global__ __launch_bounds__(256) void k_gemm_out(
        const __bf16* __restrict__ A, const __bf16* __restrict__ Bt,
        const float* __restrict__ bias, const float* __restrict__ xres,
        float* __restrict__ out) {
    constexpr int BM = 128, BN = 64, KD = 1024, STEPS = KD / 32;
    __shared__ __bf16 As[2][BM][GP];
    __shared__ __bf16 Bs[2][BN][GP];

    const int tid  = threadIdx.x;
    const int lane = tid & 31, wave = tid >> 5;
    const int lh = lane & 15, hsel = lane >> 4;
    const int m0 = blockIdx.x * BM, n0 = blockIdx.y * BN;
    const int wm = wave >> 1, wn = wave & 1;     // 4x2 waves, wave tile 32x32

    auto issue_tiles = [&](int k0, int buf) {
#pragma unroll
        for (int i = 0; i < (BM * 4) / 256; ++i) {
            int q = tid + i * 256;
            int row = q >> 2, cc = q & 3;
            cp16(&As[buf][row][cc * 8], A + (size_t)(m0 + row) * KD + k0 + cc * 8);
        }
        {
            int row = tid >> 2, cc = tid & 3;    // BN*4 = 256 chunks
            cp16(&Bs[buf][row][cc * 8], Bt + (size_t)(n0 + row) * KD + k0 + cc * 8);
        }
    };

    const v8f vz = {0.f,0.f,0.f,0.f,0.f,0.f,0.f,0.f};
    v8f acc[2][2];
#pragma unroll
    for (int i = 0; i < 2; ++i)
#pragma unroll
        for (int j = 0; j < 2; ++j) acc[i][j] = vz;

    issue_tiles(0, 0);
    for (int s = 0; s < STEPS; ++s) {
        const int cur = s & 1;
        if (s + 1 < STEPS) {
            issue_tiles((s + 1) * 32, (s + 1) & 1);
            cp_wait<3>();
        } else {
            cp_wait<0>();
        }
        __syncthreads();

        Frag16 af[2];
#pragma unroll
        for (int mt = 0; mt < 2; ++mt) {
            const __bf16* p = &As[cur][wm * 32 + mt * 16 + lh][hsel * 8];
            af[mt].q[0] = *(const uint4*)p;
            af[mt].q[1] = *(const uint4*)(p + 16);
        }
#pragma unroll
        for (int nt = 0; nt < 2; ++nt) {
            Frag16 bfr;
            const __bf16* p = &Bs[cur][wn * 32 + nt * 16 + lh][hsel * 8];
            bfr.q[0] = *(const uint4*)p;
            bfr.q[1] = *(const uint4*)(p + 16);
#pragma unroll
            for (int mt = 0; mt < 2; ++mt)
                acc[mt][nt] = wmma_bf16(af[mt].v, bfr.v, acc[mt][nt]);
        }
        __syncthreads();
    }

#pragma unroll
    for (int mt = 0; mt < 2; ++mt)
#pragma unroll
        for (int nt = 0; nt < 2; ++nt)
#pragma unroll
            for (int r = 0; r < 8; ++r) {
                int m = m0 + wm * 32 + mt * 16 + r + hsel * 8;
                int n = n0 + wn * 32 + nt * 16 + lh;
                out[(size_t)m * 256 + n] =
                    acc[mt][nt][r] + bias[n] + xres[(size_t)m * 256 + n];
            }
}

// ---------------------------------------------------------------------------
// Host launch
// ---------------------------------------------------------------------------
extern "C" void kernel_launch(void* const* d_in, const int* in_sizes, int n_in,
                              void* d_out, int out_size, void* d_ws, size_t ws_size,
                              hipStream_t stream) {
    (void)in_sizes; (void)n_in; (void)out_size; (void)ws_size;

    const float* x    = (const float*)d_in[0];   // [8,32,32,256]
    const float* Wqkv = (const float*)d_in[1];   // [256,3072]
    const float* Wout = (const float*)d_in[2];   // [1024,256]
    const float* bout = (const float*)d_in[3];   // [256]
    float* out = (float*)d_out;

    char* ws = (char*)d_ws;
    size_t off = 0;
    auto alloc = [&](size_t bytes) -> void* {
        void* p = ws + off;
        off += (bytes + 255) & ~(size_t)255;
        return p;
    };

    __bf16* Xb     = (__bf16*)alloc((size_t)8192 * 256 * 2);
    __bf16* Wqkv_t = (__bf16*)alloc((size_t)3072 * 256 * 2);
    __bf16* Wout_t = (__bf16*)alloc((size_t)256 * 1024 * 2);
    __bf16* Qp     = (__bf16*)alloc((size_t)32 * 1024 * 256 * 2);
    __bf16* Kp     = (__bf16*)alloc((size_t)32 * 1024 * 256 * 2);
    __bf16* Vtp    = (__bf16*)alloc((size_t)32 * 256 * 1024 * 2);
    __bf16* Rp     = (__bf16*)alloc((size_t)8192 * 1024 * 2);

    k_cvt_bf16<<<8192, 256, 0, stream>>>(x, Xb, 8192 * 256);
    k_transpose_bf16<<<(256 * 3072) / 256, 256, 0, stream>>>(Wqkv, Wqkv_t, 256, 3072);
    k_transpose_bf16<<<(1024 * 256) / 256, 256, 0, stream>>>(Wout, Wout_t, 1024, 256);

    k_gemm_qkv<<<dim3(64, 24), 256, 0, stream>>>(Xb, Wqkv_t, Qp, Kp, Vtp);
    k_attention<<<dim3(8, 4, 8), 256, 0, stream>>>(Qp, Kp, Vtp, Rp);
    k_gemm_out<<<dim3(64, 4), 256, 0, stream>>>(Rp, Wout_t, bout, x, out);
}
